// BiLSTMLayer_24867860644483
// MI455X (gfx1250) — compile-verified, matrix operand order checked by reading, not run on hardware
//
#include <hip/hip_runtime.h>
#include <hip/hip_bf16.h>
#include <math.h>

// ---------------- problem constants ----------------
constexpr int Bn = 32;          // batch
constexpr int Tn = 512;         // time
constexpr int In = 769;         // input features
constexpr int Hn = 300;         // hidden
constexpr int G4 = 4 * Hn;      // 1200 gate columns
constexpr int Mn = Bn * Tn;     // 16384 rows of the input-projection GEMM

constexpr int KP  = 800;        // x / W_ih K padded to 800 (25 chunks of 32)
constexpr int NB  = 1280;       // W_ih rows padded to 1280 (10 tiles of 128)
constexpr int KC1 = KP / 32;    // 25
constexpr int NT2 = G4 / 16;    // 75 N-tiles of 16 for recurrence
constexpr int KC2 = 10;         // recurrence K padded 300 -> 320, 10 chunks of 32

constexpr int AST = 40;         // LDS tile K-stride (bf16), 80B rows: aligned, skewed banks
constexpr int GST = 1204;       // gates LDS row stride (f32)
constexpr int HS  = 328;        // h LDS row stride (bf16), >= 320, 16B-aligned rows

// ---------------- WMMA vector types ----------------
typedef __attribute__((ext_vector_type(16))) __bf16 v16bf;
typedef __attribute__((ext_vector_type(8)))  __bf16 v8bf;
typedef __attribute__((ext_vector_type(8)))  float  v8f;

__device__ __forceinline__ v8f wmma_bf16(v16bf a, v16bf b, v8f c) {
  // (neg_a, A, neg_b, B, c_mod, C, reuse_a, reuse_b)
  return __builtin_amdgcn_wmma_f32_16x16x32_bf16(false, a, false, b, (short)0, c,
                                                 false, false);
}

// Per-lane 16x32 fragment row from an LDS row (bf16, k-major).
// Lanes 0-15: K {0..7, 16..23}; lanes 16-31: K {8..15, 24..31} (ISA 7.12.2).
__device__ __forceinline__ v16bf frag_ld(const __bf16* rowp, int kb) {
  union U { struct { v8bf lo, hi; } s; v16bf v; } u;
  u.s.lo = *(const v8bf*)(rowp + kb);
  u.s.hi = *(const v8bf*)(rowp + kb + 16);
  return u.v;
}

// ============================================================
// Prep kernels: fp32 -> bf16 with zero padding
// ============================================================
__global__ __launch_bounds__(256) void bilstm_cvt_x(const float* __restrict__ x,
                                                    __bf16* __restrict__ xb) {
  size_t e = (size_t)blockIdx.x * 256 + threadIdx.x;   // over Mn*KP
  if (e >= (size_t)Mn * KP) return;
  int m = (int)(e / KP), k = (int)(e % KP);
  float v = (k < In) ? x[(size_t)m * In + k] : 0.0f;
  xb[e] = (__bf16)v;
}

__global__ __launch_bounds__(256) void bilstm_cvt_w(const float* __restrict__ Wf,
                                                    const float* __restrict__ Wb,
                                                    __bf16* __restrict__ out) {
  const int dir = blockIdx.y;
  size_t e = (size_t)blockIdx.x * 256 + threadIdx.x;   // over NB*KP
  if (e >= (size_t)NB * KP) return;
  int g = (int)(e / KP), k = (int)(e % KP);
  const float* W = dir ? Wb : Wf;
  float v = (g < G4 && k < In) ? W[(size_t)g * In + k] : 0.0f;
  out[(size_t)dir * NB * KP + e] = (__bf16)v;
}

// ============================================================
// Repack W_hh [1200,300] f32 -> bf16 WMMA B-fragments
// layout: [dir][nt*KC2+kc][lane][16 bf16], K zero-padded to 320
// ============================================================
__global__ void bilstm_pack_whh(const float* __restrict__ Wf,
                                const float* __restrict__ Wb,
                                __bf16* __restrict__ out) {
  const int dir  = blockIdx.y;
  const int g    = blockIdx.x;            // nt*KC2 + kc
  const int kc   = g % KC2;
  const int nt   = g / KC2;
  const int lane = threadIdx.x;           // 0..31
  const float* W = dir ? Wb : Wf;

  const int n  = nt * 16 + (lane & 15);
  const int kb = kc * 32 + ((lane < 16) ? 0 : 8);

  union U { __bf16 e[16]; v16bf v; } u;
#pragma unroll
  for (int j = 0; j < 16; ++j) {
    int k = kb + (j < 8 ? j : j + 8);
    float val = (k < Hn) ? W[(size_t)n * Hn + k] : 0.0f;
    u.e[j] = (__bf16)val;
  }
  ((v16bf*)out)[((size_t)(dir * (NT2 * KC2) + g) * 32 + lane)] = u.v;
}

// ============================================================
// xp = x @ W_ih^T + b  (both dirs).  WG tile 128x128, 8 waves of
// 32x64 (2x4 WMMA tiles), bf16 inputs, double-buffered LDS.
// ============================================================
__global__ __launch_bounds__(256) void bilstm_xp_gemm(
    const __bf16* __restrict__ xb, const __bf16* __restrict__ wb,
    const float* __restrict__ bf_, const float* __restrict__ bb_,
    float* __restrict__ xp) {
  __shared__ __attribute__((aligned(16))) __bf16 sA[2][128 * AST];
  __shared__ __attribute__((aligned(16))) __bf16 sB[2][128 * AST];

  const int tid  = threadIdx.x;
  const int lane = tid & 31;
  const int wave = tid >> 5;
  const int dir  = blockIdx.z;
  const float*  bias = dir ? bb_ : bf_;
  const __bf16* Wd   = wb + (size_t)dir * NB * KP;
  float* xpd = xp + (size_t)dir * Mn * G4;

  const int Mbase = blockIdx.x * 128;
  const int Nbase = blockIdx.y * 128;
  const int wm = (wave & 3) * 32;    // wave sub-tile M offset
  const int wn = (wave >> 2) * 64;   // wave sub-tile N offset

  const int kb = (lane < 16) ? 0 : 8;
  const int lr = lane & 15;

  // staging map: 128x32 tile = 512 groups of 8 bf16; thread owns groups tid, tid+256
  const int ra0 = tid >> 2,         ca0 = (tid & 3) * 8;
  const int ra1 = (tid + 256) >> 2, ca1 = (tid & 3) * 8;

  v8f acc[2][4];
#pragma unroll
  for (int i = 0; i < 2; ++i)
#pragma unroll
    for (int j = 0; j < 4; ++j)
      acc[i][j] = (v8f){0.f, 0.f, 0.f, 0.f, 0.f, 0.f, 0.f, 0.f};

  const __bf16* arow0 = xb + (size_t)(Mbase + ra0) * KP + ca0;
  const __bf16* arow1 = xb + (size_t)(Mbase + ra1) * KP + ca1;
  const __bf16* brow0 = Wd + (size_t)(Nbase + ra0) * KP + ca0;
  const __bf16* brow1 = Wd + (size_t)(Nbase + ra1) * KP + ca1;

  // prologue: stage chunk 0 into buffer 0
  {
    v8bf A0 = *(const v8bf*)(arow0), A1 = *(const v8bf*)(arow1);
    v8bf B0 = *(const v8bf*)(brow0), B1 = *(const v8bf*)(brow1);
    *(v8bf*)&sA[0][ra0 * AST + ca0] = A0;
    *(v8bf*)&sA[0][ra1 * AST + ca1] = A1;
    *(v8bf*)&sB[0][ra0 * AST + ca0] = B0;
    *(v8bf*)&sB[0][ra1 * AST + ca1] = B1;
  }
  __syncthreads();

  for (int kc = 0; kc < KC1; ++kc) {
    const int buf  = kc & 1;
    const bool more = (kc + 1) < KC1;
    v8bf nA0, nA1, nB0, nB1;
    if (more) {  // prefetch next chunk (hides under WMMAs below)
      int k1 = (kc + 1) * 32;
      nA0 = *(const v8bf*)(arow0 + k1);
      nA1 = *(const v8bf*)(arow1 + k1);
      nB0 = *(const v8bf*)(brow0 + k1);
      nB1 = *(const v8bf*)(brow1 + k1);
    }

    v16bf a0 = frag_ld(&sA[buf][(wm + lr) * AST], kb);
    v16bf a1 = frag_ld(&sA[buf][(wm + 16 + lr) * AST], kb);
    v16bf bfr[4];
#pragma unroll
    for (int j = 0; j < 4; ++j)
      bfr[j] = frag_ld(&sB[buf][(wn + j * 16 + lr) * AST], kb);

#pragma unroll
    for (int j = 0; j < 4; ++j) {
      acc[0][j] = wmma_bf16(a0, bfr[j], acc[0][j]);
      acc[1][j] = wmma_bf16(a1, bfr[j], acc[1][j]);
    }

    if (more) {
      *(v8bf*)&sA[buf ^ 1][ra0 * AST + ca0] = nA0;
      *(v8bf*)&sA[buf ^ 1][ra1 * AST + ca1] = nA1;
      *(v8bf*)&sB[buf ^ 1][ra0 * AST + ca0] = nB0;
      *(v8bf*)&sB[buf ^ 1][ra1 * AST + ca1] = nB1;
    }
    __syncthreads();
  }

  // store D tiles: VGPR v holds M = v / v+8 (by lane half), N = lane&15
  const int ro = (lane < 16) ? 0 : 8;
#pragma unroll
  for (int i = 0; i < 2; ++i)
#pragma unroll
    for (int j = 0; j < 4; ++j) {
      int col = Nbase + wn + j * 16 + lr;
      if (col < G4) {
        float bv = bias[col];
        int mrow = Mbase + wm + i * 16 + ro;
#pragma unroll
        for (int v = 0; v < 8; ++v)
          xpd[(size_t)(mrow + v) * G4 + col] = acc[i][j][v] + bv;
      }
    }
}

// ============================================================
// Persistent recurrence, one WG per direction.
// LDS: gates[32][1204] f32, h[32][328] bf16 (K-padded), c[32][300] f32.
// ============================================================
__global__ __launch_bounds__(256) void bilstm_recurrence(
    const float* __restrict__ xp, const __bf16* __restrict__ wpk,
    float* __restrict__ out) {
  extern __shared__ char smem[];
  float*  gates = (float*)smem;                                   // 32*GST f32
  __bf16* hbuf  = (__bf16*)(smem + (size_t)32 * GST * 4);         // 32*HS  bf16
  float*  cbuf  = (float*)(smem + (size_t)32 * GST * 4 + (size_t)32 * HS * 2);

  const int dir  = blockIdx.x;
  const int tid  = threadIdx.x;
  const int lane = tid & 31;
  const int wave = tid >> 5;

  const float* xpd = xp + (size_t)dir * Mn * G4;
  const v16bf* wfr = (const v16bf*)wpk + (size_t)dir * NT2 * KC2 * 32;
  float* ret = out + Bn * 2 * Hn;   // hs_f after hid_state
  float* hid = out;

  for (int e = tid; e < 32 * HS; e += 256) hbuf[e] = (__bf16)0.0f;
  for (int e = tid; e < 32 * Hn; e += 256) cbuf[e] = 0.0f;
  __syncthreads();

  const int kb = (lane < 16) ? 0 : 8;
  const int lr = lane & 15;
  const int ro = (lane < 16) ? 0 : 8;

  for (int ti = 0; ti < Tn; ++ti) {
    const int t = dir ? (Tn - 1 - ti) : ti;

    // ---- gates = h @ W_hh^T + xp[t]  (wave-uniform tile loop) ----
    for (int nt = wave; nt < NT2; nt += 8) {
      const int col = nt * 16 + lr;

      // issue the strided xp loads first so they overlap the WMMAs
      float xv0[8], xv1[8];
#pragma unroll
      for (int v = 0; v < 8; ++v) {
        xv0[v] = xpd[((size_t)(ro + v) * Tn + t) * G4 + col];
        xv1[v] = xpd[((size_t)(16 + ro + v) * Tn + t) * G4 + col];
      }

      v8f a0 = (v8f){0.f, 0.f, 0.f, 0.f, 0.f, 0.f, 0.f, 0.f};
      v8f a1 = (v8f){0.f, 0.f, 0.f, 0.f, 0.f, 0.f, 0.f, 0.f};
      const v16bf* wnt = wfr + (size_t)nt * KC2 * 32 + lane;
#pragma unroll
      for (int kc = 0; kc < KC2; ++kc) {
        v16bf bfrag = wnt[kc * 32];
        v16bf f0 = frag_ld(hbuf + lr * HS + kc * 32, kb);
        v16bf f1 = frag_ld(hbuf + (16 + lr) * HS + kc * 32, kb);
        a0 = wmma_bf16(f0, bfrag, a0);
        a1 = wmma_bf16(f1, bfrag, a1);
      }
#pragma unroll
      for (int v = 0; v < 8; ++v) {
        gates[(ro + v) * GST + col]      = a0[v] + xv0[v];
        gates[(16 + ro + v) * GST + col] = a1[v] + xv1[v];
      }
    }
    __syncthreads();

    // ---- LSTM cell elementwise update ----
    for (int e = tid; e < Bn * Hn; e += 256) {
      int b = e / Hn, n = e % Hn;
      float ig = gates[b * GST + n];
      float fg = gates[b * GST + Hn + n];
      float gg = gates[b * GST + 2 * Hn + n];
      float og = gates[b * GST + 3 * Hn + n];
      float c  = cbuf[e];
      float si = 1.0f / (1.0f + __expf(-ig));
      float sf = 1.0f / (1.0f + __expf(-fg));
      float so = 1.0f / (1.0f + __expf(-og));
      c = sf * c + si * tanhf(gg);
      float h = so * tanhf(c);
      cbuf[e] = c;
      hbuf[b * HS + n] = (__bf16)h;
      if (dir == 0) ret[((size_t)b * Tn + t) * Hn + n] = h;        // hs_f
      if (ti == Tn - 1) hid[((size_t)b * 2 + dir) * Hn + n] = h;   // hT
    }
    __syncthreads();
  }
}

// ============================================================
extern "C" void kernel_launch(void* const* d_in, const int* in_sizes, int n_in,
                              void* d_out, int out_size, void* d_ws, size_t ws_size,
                              hipStream_t stream) {
  const float* x     = (const float*)d_in[0];
  const float* Wih_f = (const float*)d_in[1];
  const float* Whh_f = (const float*)d_in[2];
  const float* b_f   = (const float*)d_in[3];
  const float* Wih_b = (const float*)d_in[4];
  const float* Whh_b = (const float*)d_in[5];
  const float* b_b   = (const float*)d_in[6];
  float* out = (float*)d_out;

  // workspace layout (all offsets 32B-aligned):
  char* ws = (char*)d_ws;
  float*  xp   = (float*)ws;                                   // 2*Mn*G4 f32
  size_t  off  = (size_t)2 * Mn * G4 * sizeof(float);          // 157.3 MB
  __bf16* wpk  = (__bf16*)(ws + off);                          // packed W_hh frags
  off += (size_t)2 * NT2 * KC2 * 32 * 16 * sizeof(__bf16);     // 1.54 MB
  __bf16* xb   = (__bf16*)(ws + off);                          // x bf16, K-padded
  off += (size_t)Mn * KP * sizeof(__bf16);                     // 26.2 MB
  __bf16* wihb = (__bf16*)(ws + off);                          // W_ih bf16, padded

  bilstm_cvt_x<<<dim3(((size_t)Mn * KP + 255) / 256), 256, 0, stream>>>(x, xb);
  bilstm_cvt_w<<<dim3(((size_t)NB * KP + 255) / 256, 2), 256, 0, stream>>>(
      Wih_f, Wih_b, wihb);
  bilstm_pack_whh<<<dim3(NT2 * KC2, 2), 32, 0, stream>>>(Whh_f, Whh_b, wpk);

  bilstm_xp_gemm<<<dim3(Mn / 128, NB / 128, 2), 256, 0, stream>>>(
      xb, wihb, b_f, b_b, xp);

  size_t smem = (size_t)32 * GST * 4 + (size_t)32 * HS * 2 + (size_t)32 * Hn * 4;
  bilstm_recurrence<<<dim3(2), 256, smem, stream>>>(xp, wpk, out);
}
